// AutoIntMLP_52733608460343
// MI455X (gfx1250) — matrix operand order repaired and checked by reading, not compiled, not run on hardware
//
#include <hip/hip_runtime.h>
#include <hip/hip_bf16.h>
#include <math.h>

// ---------------------------------------------------------------------------
// CDNA5 (gfx1250) WMMA helpers: V_WMMA_F32_16X16X32_F16, wave32.
// Fragment layouts per CDNA5 ISA 7.12.2:
//   A (16x32 f16): lane L holds row M=L%16; halves of lane split K:
//     lanes 0-15:  elems 0..7 = K 0..7,  elems 8..15 = K 16..23
//     lanes 16-31: elems 0..7 = K 8..15, elems 8..15 = K 24..31
//   B (32x16 f16): lane L holds col N=L%16; same K split as A
//   C/D (16x16 f32): vreg r: lanes 0-15 -> (M=r, N=lane); 16-31 -> (M=r+8)
//
// All B operands are sourced from [N][K]-transposed storage so each lane reads
// two runs of 8 contiguous halfs -> two 16-byte vector loads (global_load_b128 / ds b128).
// ---------------------------------------------------------------------------
typedef __attribute__((ext_vector_type(16))) _Float16 v16h;
typedef __attribute__((ext_vector_type(8)))  _Float16 v8h;
typedef __attribute__((ext_vector_type(8)))  float    v8f;
typedef __attribute__((ext_vector_type(4)))  float    v4f;

__device__ __forceinline__ v8f wmma32(v16h a, v16h b, v8f c) {
  return __builtin_amdgcn_wmma_f32_16x16x32_f16(
      /*neg_a=*/false, a, /*neg_b=*/false, b,
      /*c_mod=*/(short)0, c, /*reuse_a=*/false, /*reuse_b=*/false);
}

// A fragment: 16x32 tile, element (m,k) = src[m*ld + k], row-major.
// Requires: src 16B-aligned, ld multiple of 8 halfs. Two b128 loads per lane.
__device__ __forceinline__ v16h load_a(const _Float16* src, int ld) {
  int lane = threadIdx.x & 31;
  int m  = lane & 15;
  int kb = (lane >> 4) * 8;
  const _Float16* p = src + m * ld + kb;
  v8h lo = *(const v8h*)p;          // K = kb .. kb+7
  v8h hi = *(const v8h*)(p + 16);   // K = kb+16 .. kb+23
  return __builtin_shufflevector(lo, hi, 0, 1, 2, 3, 4, 5, 6, 7,
                                 8, 9, 10, 11, 12, 13, 14, 15);
}

// B fragment from transposed storage: element (k,n) = src[n*ldn + k].
// Requires: src 16B-aligned, ldn multiple of 8 halfs. Two b128 loads per lane.
__device__ __forceinline__ v16h load_bt(const _Float16* src, int ldn) {
  int lane = threadIdx.x & 31;
  int n  = lane & 15;
  int kb = (lane >> 4) * 8;
  const _Float16* p = src + n * ldn + kb;
  v8h lo = *(const v8h*)p;          // K = kb .. kb+7
  v8h hi = *(const v8h*)(p + 16);   // K = kb+16 .. kb+23
  return __builtin_shufflevector(lo, hi, 0, 1, 2, 3, 4, 5, 6, 7,
                                 8, 9, 10, 11, 12, 13, 14, 15);
}

// D store, row-major f16: element (m,n) -> dst[m*ld + n] (scalar, strided)
__device__ __forceinline__ void store_d_f16(_Float16* dst, int ld, v8f c) {
  int lane = threadIdx.x & 31;
  int n  = lane & 15;
  int mb = (lane >> 4) * 8;
#pragma unroll
  for (int r = 0; r < 8; ++r) dst[(mb + r) * ld + n] = (_Float16)c[r];
}

// D store, transposed f16: element (m,n) -> dst[n*ld + m] (one b128 per lane)
__device__ __forceinline__ void store_d_f16_T(_Float16* dst, int ld, v8f c) {
  int lane = threadIdx.x & 31;
  int n  = lane & 15;
  int mb = (lane >> 4) * 8;
  v8h h;
#pragma unroll
  for (int r = 0; r < 8; ++r) h[r] = (_Float16)c[r];
  *(v8h*)(dst + n * ld + mb) = h;
}

// D store, row-major f32 (scores)
__device__ __forceinline__ void store_d_f32(float* dst, int ld, v8f c) {
  int lane = threadIdx.x & 31;
  int n  = lane & 15;
  int mb = (lane >> 4) * 8;
#pragma unroll
  for (int r = 0; r < 8; ++r) dst[(mb + r) * ld + n] = c[r];
}

// ---------------------------------------------------------------------------
// Model constants
// ---------------------------------------------------------------------------
#define NUM_FIELDS 39
#define EMB        64
#define NPAD       48          // 39 tokens padded to 3 WMMA M-tiles
#define BATCH      16384
#define FLATD      (NUM_FIELDS * EMB)   // 2496
#define SSTRIDE    (NPAD * EMB)         // 3072 f16 per padded sample

// ---------------------------------------------------------------------------
// Kernel: f32 -> f16 transpose-convert. dst[n*K + k] = src[k*N + n].
// (makes every WMMA B operand contiguous per lane)
// ---------------------------------------------------------------------------
__global__ void tr_cvt_kernel(const float* __restrict__ src,
                              _Float16* __restrict__ dst, int K, int N) {
  int i = blockIdx.x * blockDim.x + threadIdx.x;
  if (i >= K * N) return;
  int n = i / K, k = i - n * K;
  dst[(size_t)n * K + k] = (_Float16)src[(size_t)k * N + n];
}

// ---------------------------------------------------------------------------
// Kernel: embedding gather -> padded f16 activations (B x 48 x 64).
// 16-byte granularity: two v4f global loads -> one v8h store per chunk.
// ---------------------------------------------------------------------------
__global__ void gather_kernel(const int* __restrict__ X,
                              const float* __restrict__ emb,
                              _Float16* __restrict__ E16) {
  int b = blockIdx.x;
  for (int i = threadIdx.x; i < NPAD * 8; i += blockDim.x) {  // 8 halfs / chunk
    int f = i >> 3, q = i & 7;
    v8h h = {};
    if (f < NUM_FIELDS) {
      int row = X[(size_t)b * NUM_FIELDS + f] + f * 1000;
      const float* rp = emb + (size_t)row * EMB + q * 8;
      v4f f0 = *(const v4f*)rp;
      v4f f1 = *(const v4f*)(rp + 4);
#pragma unroll
      for (int j = 0; j < 4; ++j) { h[j] = (_Float16)f0[j]; h[j + 4] = (_Float16)f1[j]; }
    }
    *(v8h*)(E16 + (size_t)b * SSTRIDE + (size_t)i * 8) = h;
  }
}

// ---------------------------------------------------------------------------
// Kernel: fused 3-layer MHSA per sample + Wlin logit.
// 1 block (4 waves / 128 threads) per sample; all tiles live in LDS.
// Weights arrive transposed ([out][in]); V is kept transposed in LDS.
// ---------------------------------------------------------------------------
__global__ __launch_bounds__(128)
void attn_kernel(const _Float16* __restrict__ E16,
                 const _Float16* __restrict__ WQ16,   // [l][n][k] 64x64 each
                 const _Float16* __restrict__ WK16,
                 const _Float16* __restrict__ WV16,
                 const _Float16* __restrict__ WR16,
                 const float* __restrict__ Wlin,
                 float* __restrict__ att_out) {
  __shared__ __align__(16) _Float16 xb[NPAD * EMB];      // activations (6 KB)
  __shared__ __align__(16) _Float16 qb[NPAD * EMB];      // Q           (6 KB)
  __shared__ __align__(16) _Float16 kb[NPAD * EMB];      // K           (6 KB)
  __shared__ __align__(16) _Float16 vb[EMB * 64];        // V^T [dim][tok] (8 KB)
  __shared__ __align__(16) _Float16 rb[NPAD * EMB];      // x @ Wr      (6 KB)
  __shared__ __align__(16) float    sb[2 * NPAD * NPAD]; // raw scores  (18 KB)
  __shared__ __align__(16) _Float16 ab[2 * NPAD * 64];   // softmax     (12 KB)
  __shared__ float red[128];

  const int b    = blockIdx.x;
  const int tid  = threadIdx.x;
  const int wave = tid >> 5;

  {
    const v8h* src = (const v8h*)(E16 + (size_t)b * SSTRIDE);
    v8h* dx = (v8h*)xb;
    for (int i = tid; i < SSTRIDE / 8; i += 128) dx[i] = src[i];
    v8h* dv = (v8h*)vb;
    v8h z = {};
    for (int i = tid; i < (EMB * 64) / 8; i += 128) dv[i] = z;  // tok 48..63 stay 0
  }
  __syncthreads();

  for (int l = 0; l < 3; ++l) {
    // ---- projections: Q,K,V,R = x @ W  (16 tasks = 4 mats x 4 N-tiles) ----
    for (int t = wave; t < 16; t += 4) {
      int ws = t >> 2, nt = t & 3;
      const _Float16* Wsel =
          (ws == 0) ? (WQ16 + l * 4096) :
          (ws == 1) ? (WK16 + l * 4096) :
          (ws == 2) ? (WV16 + l * 4096) : (WR16 + l * 4096);
      for (int mt = 0; mt < 3; ++mt) {
        v8f acc = {};
#pragma unroll
        for (int ks = 0; ks < 2; ++ks) {
          v16h a  = load_a(xb + mt * 16 * EMB + ks * 32, EMB);
          v16h bb = load_bt(Wsel + nt * 16 * EMB + ks * 32, EMB);
          acc = wmma32(a, bb, acc);
        }
        if (ws == 2) {
          store_d_f16_T(vb + nt * 16 * 64 + mt * 16, 64, acc);  // V^T
        } else {
          _Float16* Dsel = (ws == 0) ? qb : (ws == 1) ? kb : rb;
          store_d_f16(Dsel + mt * 16 * EMB + nt * 16, EMB, acc);
        }
      }
    }
    __syncthreads();

    // ---- scores: S_h = Q_h @ K_h^T  (6 tasks = 2 heads x 3 M-tiles) ----
    // B operand: element (d, j) = kb[j*64 + h*32 + d]  (contiguous in d)
    for (int t = wave; t < 6; t += 4) {
      int h = t / 3, mt = t % 3;
      for (int nt = 0; nt < 3; ++nt) {
        v8f acc = {};
        v16h a  = load_a(qb + mt * 16 * EMB + h * 32, EMB);
        v16h bb = load_bt(kb + (nt * 16) * EMB + h * 32, EMB);
        acc = wmma32(a, bb, acc);
        store_d_f32(sb + h * NPAD * NPAD + mt * 16 * NPAD + nt * 16, NPAD, acc);
      }
    }
    __syncthreads();

    // ---- softmax over j<39; pad cols/rows -> 0 so padded tiles are inert ----
    for (int row = tid; row < 2 * NPAD; row += 128) {
      int h = row / NPAD, i = row % NPAD;
      _Float16* arow = ab + h * NPAD * 64 + i * 64;
      if (i < NUM_FIELDS) {
        const float* srow = sb + h * NPAD * NPAD + i * NPAD;
        float mx = srow[0];
        for (int j = 1; j < NUM_FIELDS; ++j) mx = fmaxf(mx, srow[j]);
        float sum = 0.0f;
        for (int j = 0; j < NUM_FIELDS; ++j) sum += __expf(srow[j] - mx);
        float inv = 1.0f / sum;
        for (int j = 0; j < NUM_FIELDS; ++j)
          arow[j] = (_Float16)(__expf(srow[j] - mx) * inv);
        for (int j = NUM_FIELDS; j < 64; ++j) arow[j] = (_Float16)0.0f;
      } else {
        v8h z = {};
        for (int j = 0; j < 8; ++j) ((v8h*)arow)[j] = z;
      }
    }
    __syncthreads();

    // ---- out = attn @ V_h, += residual proj, ReLU -> xb ----
    // 12 tasks = 2 heads x 3 M-tiles x 2 N-tiles(dh=32)
    // B operand from V^T: element (tok k, dim n) = vb[(h*32+nt*16+n)*64 + k]
    for (int t = wave; t < 12; t += 4) {
      int h = t / 6, rem = t % 6, mt = rem >> 1, nt = rem & 1;
      v8f acc = {};
#pragma unroll
      for (int ks = 0; ks < 2; ++ks) {
        v16h a  = load_a(ab + h * NPAD * 64 + mt * 16 * 64 + ks * 32, 64);
        v16h bb = load_bt(vb + (h * 32 + nt * 16) * 64 + ks * 32, 64);
        acc = wmma32(a, bb, acc);
      }
      int lane = tid & 31, n = lane & 15, mb = (lane >> 4) * 8;
#pragma unroll
      for (int r = 0; r < 8; ++r) {
        int m  = mt * 16 + mb + r;
        int nc = h * 32 + nt * 16 + n;
        float v = acc[r] + (float)rb[m * EMB + nc];
        xb[m * EMB + nc] = (_Float16)(v > 0.0f ? v : 0.0f);
      }
    }
    __syncthreads();
  }

  // ---- att_logit = relu( flatten(att) . Wlin ) ----
  float part = 0.0f;
  for (int i = tid; i < FLATD; i += 128) part += (float)xb[i] * Wlin[i];
  red[tid] = part;
  __syncthreads();
  for (int s = 64; s > 0; s >>= 1) {
    if (tid < s) red[tid] += red[tid + s];
    __syncthreads();
  }
  if (tid == 0) att_out[b] = fmaxf(red[0], 0.0f);
}

// ---------------------------------------------------------------------------
// Kernel: WMMA GEMM C = relu(A @ B^T + bias); B given transposed [N][K].
// Block = 256 threads (8 waves); block tile M=128 x N=128.
// Wave w owns rows w*16..+16 and all 8 N-subtiles (8 v8f accumulators).
// Halves A re-reads vs an N=64 tile and gives 8 WMMA per 18 b128 loads.
// ---------------------------------------------------------------------------
__global__ __launch_bounds__(256)
void gemm_bias_relu_kernel(const _Float16* __restrict__ A, int lda,
                           const _Float16* __restrict__ BT,   // [N][K]
                           const float* __restrict__ bias,
                           _Float16* __restrict__ C, int ldc, int K) {
  int wave = threadIdx.x >> 5;
  int m0 = blockIdx.x * 128 + wave * 16;
  int n0 = blockIdx.y * 128;

  v8f acc[8] = {};
  for (int k = 0; k < K; k += 32) {
    const _Float16* ap = A + (size_t)m0 * lda + k;
    if (k + 32 < K) __builtin_prefetch(ap + 32, 0, 0);   // global_prefetch_b8
    v16h a = load_a(ap, lda);
#pragma unroll
    for (int nt = 0; nt < 8; ++nt) {
      v16h bb = load_bt(BT + (size_t)(n0 + nt * 16) * K + k, K);
      acc[nt] = wmma32(a, bb, acc[nt]);
    }
  }

  int lane = threadIdx.x & 31;
  int n = lane & 15, mb = (lane >> 4) * 8;
#pragma unroll
  for (int nt = 0; nt < 8; ++nt) {
    int nc = n0 + nt * 16 + n;
    float bv = bias[nc];
#pragma unroll
    for (int r = 0; r < 8; ++r) {
      float v = acc[nt][r] + bv;
      C[(size_t)(m0 + mb + r) * ldc + nc] = (_Float16)(v > 0.0f ? v : 0.0f);
    }
  }
}

// ---------------------------------------------------------------------------
// Kernel: out = sigmoid( relu(h2 @ W3 + b3) + att_logit )
// ---------------------------------------------------------------------------
__global__ void final_kernel(const _Float16* __restrict__ H2,
                             const float* __restrict__ W3,
                             const float* __restrict__ b3,
                             const float* __restrict__ att,
                             float* __restrict__ out, int n) {
  int i = blockIdx.x * blockDim.x + threadIdx.x;
  if (i >= n) return;
  const _Float16* h = H2 + (size_t)i * 256;
  float s = 0.0f;
  for (int k = 0; k < 256; ++k) s += (float)h[k] * W3[k];
  s += b3[0];
  s = s > 0.0f ? s : 0.0f;
  float z = s + att[i];
  out[i] = 1.0f / (1.0f + __expf(-z));
}

// ---------------------------------------------------------------------------
// Host launcher
// ---------------------------------------------------------------------------
extern "C" void kernel_launch(void* const* d_in, const int* in_sizes, int n_in,
                              void* d_out, int out_size, void* d_ws, size_t ws_size,
                              hipStream_t stream) {
  const int*   X    = (const int*)  d_in[0];
  const float* emb  = (const float*)d_in[1];
  const float* WQ   = (const float*)d_in[2];
  const float* WK   = (const float*)d_in[3];
  const float* WV   = (const float*)d_in[4];
  const float* WR   = (const float*)d_in[5];
  const float* W1   = (const float*)d_in[6];
  const float* b1   = (const float*)d_in[7];
  const float* W2   = (const float*)d_in[8];
  const float* b2   = (const float*)d_in[9];
  const float* W3   = (const float*)d_in[10];
  const float* b3   = (const float*)d_in[11];
  const float* Wlin = (const float*)d_in[12];
  float* out = (float*)d_out;

  // ---- workspace carve (total ~129 MB) ----
  char* ws = (char*)d_ws;
  size_t off = 0;
  auto carve = [&](size_t bytes) -> char* {
    char* p = ws + off;
    off += (bytes + 255) & ~(size_t)255;
    return p;
  };
  _Float16* E16  = (_Float16*)carve((size_t)BATCH * SSTRIDE * 2);   // 96 MB
  _Float16* WQ16 = (_Float16*)carve(3 * 4096 * 2);
  _Float16* WK16 = (_Float16*)carve(3 * 4096 * 2);
  _Float16* WV16 = (_Float16*)carve(3 * 4096 * 2);
  _Float16* WR16 = (_Float16*)carve(3 * 4096 * 2);
  _Float16* W1T  = (_Float16*)carve((size_t)FLATD * 512 * 2);
  _Float16* W2T  = (_Float16*)carve((size_t)512 * 256 * 2);
  float*    ATT  = (float*)   carve((size_t)BATCH * 4);
  _Float16* H1   = (_Float16*)carve((size_t)BATCH * 512 * 2);
  _Float16* H2   = (_Float16*)carve((size_t)BATCH * 256 * 2);

  // ---- weight transpose-convert to f16 [N][K] ----
  for (int l = 0; l < 3; ++l) {
    tr_cvt_kernel<<<(4096 + 255) / 256, 256, 0, stream>>>(WQ + l * 4096, WQ16 + l * 4096, 64, 64);
    tr_cvt_kernel<<<(4096 + 255) / 256, 256, 0, stream>>>(WK + l * 4096, WK16 + l * 4096, 64, 64);
    tr_cvt_kernel<<<(4096 + 255) / 256, 256, 0, stream>>>(WV + l * 4096, WV16 + l * 4096, 64, 64);
    tr_cvt_kernel<<<(4096 + 255) / 256, 256, 0, stream>>>(WR + l * 4096, WR16 + l * 4096, 64, 64);
  }
  tr_cvt_kernel<<<(FLATD * 512 + 255) / 256, 256, 0, stream>>>(W1, W1T, FLATD, 512);
  tr_cvt_kernel<<<(512 * 256 + 255) / 256, 256, 0, stream>>>(W2, W2T, 512, 256);

  // ---- embedding gather (padded f16, 16B-granular) ----
  gather_kernel<<<BATCH, 256, 0, stream>>>(X, emb, E16);

  // ---- fused attention stack + Wlin logit ----
  attn_kernel<<<BATCH, 128, 0, stream>>>(E16, WQ16, WK16, WV16, WR16, Wlin, ATT);

  // ---- DNN tower ----
  gemm_bias_relu_kernel<<<dim3(BATCH / 128, 512 / 128), 256, 0, stream>>>(
      E16, SSTRIDE, W1T, b1, H1, 512, FLATD);
  gemm_bias_relu_kernel<<<dim3(BATCH / 128, 256 / 128), 256, 0, stream>>>(
      H1, 512, W2T, b2, H2, 256, 512);

  // ---- combine + sigmoid ----
  final_kernel<<<(BATCH + 255) / 256, 256, 0, stream>>>(H2, W3, b3, ATT, out, BATCH);
}
